// SpatialRelationModuleMultiScale_7868380086930
// MI455X (gfx1250) — matrix operand order repaired and checked by reading, not compiled, not run on hardware
//
#include <hip/hip_runtime.h>
#include <stdint.h>

typedef __attribute__((ext_vector_type(16))) __bf16 v16bf;
typedef __attribute__((ext_vector_type(8)))  __bf16 v8bf;
typedef __attribute__((ext_vector_type(8)))  float  v8f;

#define B_    2048
#define G_    8
#define FEAT_ 512
#define IMG_  1024
#define K_    4096   // G_ * FEAT_
#define NSC   7
#define OUTC  7168   // NSC * IMG_

__device__ __forceinline__ uint16_t f32_to_bf16_rne(float f) {
  uint32_t u = __float_as_uint(f);
  u += 0x7fffu + ((u >> 16) & 1u);
  return (uint16_t)(u >> 16);
}

// ---------------- input f32 -> bf16 ----------------
__global__ void convert_bf16_kernel(const float* __restrict__ in,
                                    uint16_t* __restrict__ out, int n) {
  int i = blockIdx.x * blockDim.x + threadIdx.x;
  if (i < n) out[i] = f32_to_bf16_rne(in[i]);
}

// ---------------- fold perms+mean into effective weights (bf16) ----------------
// Weff[d, g*512+f] = (1/n_sub) * sum_{(s,j): perms[s,j]==g} W[d, j*512+f]
__global__ void build_weff_kernel(const float* __restrict__ W,     // [IMG_, scale*FEAT_]
                                  const int*   __restrict__ perms, // [n_sub, scale]
                                  uint16_t*    __restrict__ Weff,  // [IMG_, K_]
                                  int scale, int n_sub) {
  __shared__ float Msh[G_ * G_];  // M[g][j] weights
  int tid = threadIdx.x;
  if (tid < G_ * G_) {
    int g = tid >> 3, j = tid & 7;
    float c = 0.f;
    if (j < scale) {
      for (int s = 0; s < n_sub; ++s)
        if (perms[s * scale + j] == g) c += 1.f;
    }
    Msh[tid] = c / (float)n_sub;
  }
  __syncthreads();
  size_t idx = (size_t)blockIdx.x * blockDim.x + tid;  // over IMG_*K_
  if (idx >= (size_t)IMG_ * K_) return;
  int d = (int)(idx >> 12);          // /K_
  int k = (int)(idx & (K_ - 1));
  int g = k >> 9;                    // /FEAT_
  int f = k & (FEAT_ - 1);
  int Ks = scale * FEAT_;
  float acc = 0.f;
  for (int j = 0; j < scale; ++j)
    acc += Msh[g * G_ + j] * W[(size_t)d * Ks + j * FEAT_ + f];
  Weff[idx] = f32_to_bf16_rne(acc);
}

// ---------------- bf16 WMMA GEMM with fused bias ----------------
struct GemmParams {
  const uint16_t* Wb[NSC];
  const float*    bias[NSC];
  int             col[NSC];
};

__device__ __forceinline__ v16bf load_frag(const uint16_t* p, int gap) {
  v8bf lo = *reinterpret_cast<const v8bf*>(p);
  v8bf hi = *reinterpret_cast<const v8bf*>(p + gap);
  return __builtin_shufflevector(lo, hi, 0, 1, 2, 3, 4, 5, 6, 7,
                                 8, 9, 10, 11, 12, 13, 14, 15);
}

// Block: 128 threads = 4 waves; block tile 64(M) x 128(N); wave tile 32x64.
// Grid: (IMG_/128, B_/64, nscales). Exact fit -> EXEC all ones for WMMA.
__global__ __launch_bounds__(128)
void gemm_bf16_kernel(const uint16_t* __restrict__ A,   // [B_, K_] bf16
                      float* __restrict__ out,          // [B_, OUTC]
                      GemmParams p) {
  const uint16_t* __restrict__ Wb = p.Wb[blockIdx.z];
  const float* __restrict__ bias  = p.bias[blockIdx.z];
  const int col_off               = p.col[blockIdx.z];

  const int tid   = threadIdx.x;
  const int lane  = tid & 31;
  const int wave  = tid >> 5;
  const int row16 = lane & 15;
  const int khalf = lane >> 4;

  const int m_base = blockIdx.y * 64 + (wave & 1) * 32;
  const int n_base = blockIdx.x * 128 + (wave >> 1) * 64;

  // A frag (16x32 bf16): lane holds row M=row16, K = khalf*8 + {0..7, 16..23}
  const uint16_t* aptr[2];
#pragma unroll
  for (int ti = 0; ti < 2; ++ti)
    aptr[ti] = A + (size_t)(m_base + ti * 16 + row16) * K_ + khalf * 8;
  // B frag (32x16 bf16): lane holds col N=row16, K = khalf*8 + {0..7, 16..23}
  const uint16_t* bptr[4];
#pragma unroll
  for (int tj = 0; tj < 4; ++tj)
    bptr[tj] = Wb + (size_t)(n_base + tj * 16 + row16) * K_ + khalf * 8;

  v8f acc[2][4];
  const v8f zero8 = {0.f, 0.f, 0.f, 0.f, 0.f, 0.f, 0.f, 0.f};
#pragma unroll
  for (int ti = 0; ti < 2; ++ti)
#pragma unroll
    for (int tj = 0; tj < 4; ++tj)
      acc[ti][tj] = zero8;

  for (int k = 0; k < K_; k += 32) {
    // WGP-scope (high-locality) prefetch of the streaming K dimension; the
    // workspace extends past each tensor so trailing prefetches stay mapped.
    __builtin_prefetch(aptr[0] + k + 512, 0, 3);
    __builtin_prefetch(bptr[0] + k + 512, 0, 3);

    v16bf afrag[2], bfrag[4];
#pragma unroll
    for (int ti = 0; ti < 2; ++ti) afrag[ti] = load_frag(aptr[ti] + k, 16);
#pragma unroll
    for (int tj = 0; tj < 4; ++tj) bfrag[tj] = load_frag(bptr[tj] + k, 16);
#pragma unroll
    for (int ti = 0; ti < 2; ++ti)
#pragma unroll
      for (int tj = 0; tj < 4; ++tj)
        acc[ti][tj] = __builtin_amdgcn_wmma_f32_16x16x32_bf16(
            false, afrag[ti], false, bfrag[tj], (short)0, acc[ti][tj],
            false, false);
  }

  // Epilogue: C/D layout: VGPR r, lane l -> (M = r + (l>>4)*8, N = l&15)
#pragma unroll
  for (int tj = 0; tj < 4; ++tj) {
    const float bv  = bias[n_base + tj * 16 + row16];
    const int   col = col_off + n_base + tj * 16 + row16;
#pragma unroll
    for (int ti = 0; ti < 2; ++ti) {
      const int mrow = m_base + ti * 16 + khalf * 8;
#pragma unroll
      for (int r = 0; r < 8; ++r)
        out[(size_t)(mrow + r) * OUTC + col] = acc[ti][tj][r] + bv;
    }
  }
}

extern "C" void kernel_launch(void* const* d_in, const int* in_sizes, int n_in,
                              void* d_out, int out_size, void* d_ws, size_t ws_size,
                              hipStream_t stream) {
  (void)n_in; (void)out_size;
  const float* input = (const float*)d_in[0];

  uint16_t* Abf = (uint16_t*)d_ws;
  const size_t Abytes = (size_t)B_ * K_ * 2;     // 16 MB
  const size_t Wbytes = (size_t)IMG_ * K_ * 2;   // 8 MB per scale
  uint16_t* WeffBase = (uint16_t*)((char*)d_ws + Abytes);

  const bool multi = ws_size >= Abytes + (size_t)NSC * Wbytes;

  {
    int n = B_ * K_;
    convert_bf16_kernel<<<(n + 255) / 256, 256, 0, stream>>>(input, Abf, n);
  }

  const size_t total = (size_t)IMG_ * K_;
  const int bblocks = (int)((total + 255) / 256);
  dim3 ggrid(IMG_ / 128, B_ / 64, 1);

  if (multi) {
    GemmParams p;
    for (int i = 0; i < NSC; ++i) {
      const int scale = G_ - i;
      uint16_t* Weff = WeffBase + (size_t)i * IMG_ * K_;
      build_weff_kernel<<<bblocks, 256, 0, stream>>>(
          (const float*)d_in[2 + 3 * i], (const int*)d_in[1 + 3 * i],
          Weff, scale, in_sizes[1 + 3 * i] / scale);
      p.Wb[i]   = Weff;
      p.bias[i] = (const float*)d_in[3 + 3 * i];
      p.col[i]  = i * IMG_;
    }
    ggrid.z = NSC;
    gemm_bf16_kernel<<<ggrid, 128, 0, stream>>>(Abf, (float*)d_out, p);
  } else {
    // Reuse a single 8 MB Weff buffer; stream order serializes build->gemm.
    for (int i = 0; i < NSC; ++i) {
      const int scale = G_ - i;
      build_weff_kernel<<<bblocks, 256, 0, stream>>>(
          (const float*)d_in[2 + 3 * i], (const int*)d_in[1 + 3 * i],
          WeffBase, scale, in_sizes[1 + 3 * i] / scale);
      GemmParams p;
      for (int j = 0; j < NSC; ++j) {
        p.Wb[j]   = WeffBase;
        p.bias[j] = (const float*)d_in[3 + 3 * i];
        p.col[j]  = i * IMG_;
      }
      gemm_bf16_kernel<<<ggrid, 128, 0, stream>>>(Abf, (float*)d_out, p);
    }
  }
}
